// MoleculeEncoder_15616501088449
// MI455X (gfx1250) — matrix operand order repaired and checked by reading, not compiled, not run on hardware
//
#include <hip/hip_runtime.h>

typedef __attribute__((ext_vector_type(16))) __bf16 v16bf;
typedef __attribute__((ext_vector_type(4)))  __bf16 v4bf;
typedef __attribute__((ext_vector_type(8)))  float  v8f;
typedef __attribute__((ext_vector_type(4)))  float  v4f;
typedef __attribute__((ext_vector_type(4)))  unsigned int u32x4;

union Frag32B {
    struct { u32x4 lo; u32x4 hi; } q;
    v16bf v;
};

// ---------------------------------------------------------------------------
// Weight convert: W fp32 [K][HOUT] row-major  ->  W_T bf16 [HOUT][Kp] (K padded)
// ---------------------------------------------------------------------------
__global__ void k_cvt_wt(const float* __restrict__ W, __bf16* __restrict__ Wt,
                         int K, int Kp, int HOUT) {
    int idx = blockIdx.x * 256 + threadIdx.x;
    if (idx >= HOUT * Kp) return;
    int n = idx / Kp, k = idx - n * Kp;
    float v = (k < K) ? W[k * HOUT + n] : 0.0f;
    Wt[idx] = (__bf16)v;
}

// ---------------------------------------------------------------------------
// Generic bf16 WMMA GEMM:  out[M,HOUT] = transform(A)[M,K] @ W_T^T + bias
// INMODE: 0 = fp32 plain            (Ap = const float*,  row width K)
//         1 = fp32 pad 16->K        (Ap = const float*,  row width 16)
//         2 = (1+eps)*h + aggr      (Ap = h fp32, A2p = aggr fp32)
//         3 = bf16, affine+relu     (Ap = const __bf16*, isc/ish len K)
//         4 = bf16 plain            (Ap = const __bf16*)
// OUTMODE bits: 1 = store fp32, 2 = store bf16, 4 = relu, 8 = BN stats
// Block = 128 threads (4 waves), 64 rows per block, 16 rows per wave.
// ---------------------------------------------------------------------------
template<int K, int HOUT, int INMODE, int OUTMODE>
__global__ __launch_bounds__(128)
void k_gemm(const void* __restrict__ Ap, const void* __restrict__ A2p,
            const float* __restrict__ isc, const float* __restrict__ ish,
            const float* __restrict__ eps_p, int layer,
            const __bf16* __restrict__ Wt, const float* __restrict__ bias,
            float* __restrict__ outF, __bf16* __restrict__ outB,
            float* __restrict__ stSum, float* __restrict__ stSq, int M)
{
    constexpr int STR = K + 8;                     // bf16 elements per LDS row (16B aligned)
    __shared__ __align__(32) __bf16 As[64 * STR];
    const int tid  = threadIdx.x;
    const int row0 = blockIdx.x * 64;

    float epv = 0.0f;
    if constexpr (INMODE == 2) epv = 1.0f + eps_p[layer];

    // ---- stage activation tile to LDS (bf16), fusing the input transform ----
    for (int e = tid; e < 64 * K; e += 128) {
        int r = e / K, k = e - r * K;
        int gr = row0 + r;
        float v = 0.0f;
        if (gr < M) {
            if constexpr (INMODE == 0) {
                v = ((const float*)Ap)[(long)gr * K + k];
            } else if constexpr (INMODE == 1) {
                v = (k < 16) ? ((const float*)Ap)[(long)gr * 16 + k] : 0.0f;
            } else if constexpr (INMODE == 2) {
                v = epv * ((const float*)Ap)[(long)gr * K + k]
                        + ((const float*)A2p)[(long)gr * K + k];
            } else if constexpr (INMODE == 3) {
                float u = (float)((const __bf16*)Ap)[(long)gr * K + k];
                v = fmaxf(u * isc[k] + ish[k], 0.0f);
            } else {
                v = (float)((const __bf16*)Ap)[(long)gr * K + k];
            }
        }
        As[r * STR + k] = (__bf16)v;
    }
    __syncthreads();

    const int wave = tid >> 5;
    const int lane = tid & 31;
    const int half = (lane < 16) ? 0 : 1;
    const int lm   = wave * 16 + (lane & 15);      // A-fragment row (local)
    const int nl   = lane & 15;                    // column within 16-wide tile
    const int mbase = row0 + wave * 16 + half * 8; // first output row of this lane

    for (int nt = 0; nt < HOUT / 16; ++nt) {
        const int n = nt * 16 + nl;
        // prefetch next column-tile of weights into cache
        if (nt + 1 < HOUT / 16)
            __builtin_prefetch(&Wt[(long)(n + 16) * K], 0, 3);
        v8f acc = {};
#pragma unroll
        for (int kt = 0; kt < K / 32; ++kt) {
            Frag32B a, b;
            const __bf16* ap = &As[lm * STR + kt * 32 + half * 8];
            a.q.lo = *(const u32x4*)(ap);
            a.q.hi = *(const u32x4*)(ap + 16);
            const __bf16* bp = &Wt[(long)n * K + kt * 32 + half * 8];
            b.q.lo = *(const u32x4*)(bp);
            b.q.hi = *(const u32x4*)(bp + 16);
            acc = __builtin_amdgcn_wmma_f32_16x16x32_bf16(
                      false, a.v, false, b.v, (short)0, acc, false, false);
        }
        // ---- epilogue: bias (+relu) (+store) (+BN stats) ----
        const float bv = bias[n];
        float psum = 0.0f, psq = 0.0f;
        if (mbase + 8 <= M) {
            // fast path: all 8 rows valid -> branch-free, immediate-offset stores
            float*  pF = (OUTMODE & 1) ? (outF + (long)mbase * HOUT + n) : nullptr;
            __bf16* pB = (OUTMODE & 2) ? (outB + (long)mbase * HOUT + n) : nullptr;
#pragma unroll
            for (int r = 0; r < 8; ++r) {
                float val = acc[r] + bv;
                if constexpr (OUTMODE & 4) val = fmaxf(val, 0.0f);
                if constexpr (OUTMODE & 8) { psum += val; psq += val * val; }
                if constexpr (OUTMODE & 1) pF[(long)r * HOUT] = val;
                if constexpr (OUTMODE & 2) pB[(long)r * HOUT] = (__bf16)val;
            }
        } else {
#pragma unroll
            for (int r = 0; r < 8; ++r) {
                int m = mbase + r;
                float val = acc[r] + bv;
                if constexpr (OUTMODE & 4) val = fmaxf(val, 0.0f);
                if (m < M) {
                    if constexpr (OUTMODE & 8) { psum += val; psq += val * val; }
                    if constexpr (OUTMODE & 1) outF[(long)m * HOUT + n] = val;
                    if constexpr (OUTMODE & 2) outB[(long)m * HOUT + n] = (__bf16)val;
                }
            }
        }
        if constexpr (OUTMODE & 8) {
            psum += __shfl_xor(psum, 16, 32);
            psq  += __shfl_xor(psq, 16, 32);
            if (lane < 16) {
                atomicAdd(&stSum[n], psum);
                atomicAdd(&stSq[n],  psq);
            }
        }
    }
}

// ---------------------------------------------------------------------------
// GINE message + scatter-add: aggr[dst] += relu(h[src] + e), one wave per edge
// ---------------------------------------------------------------------------
__global__ __launch_bounds__(256)
void k_edge_msg(const float* __restrict__ h, const __bf16* __restrict__ e,
                const int* __restrict__ ei, float* __restrict__ aggr, int E)
{
    int w = (blockIdx.x * 256 + threadIdx.x) >> 5;
    int lane = threadIdx.x & 31;
    if (w >= E) return;
    int src = ei[w];
    int dst = ei[E + w];
    int c = lane * 4;
    v4f  hv = *(const v4f*)(h + (long)src * 128 + c);
    v4bf ev = *(const v4bf*)(e + (long)w   * 128 + c);
    float* ap = aggr + (long)dst * 128 + c;
    atomicAdd(ap + 0, fmaxf(hv[0] + (float)ev[0], 0.0f));
    atomicAdd(ap + 1, fmaxf(hv[1] + (float)ev[1], 0.0f));
    atomicAdd(ap + 2, fmaxf(hv[2] + (float)ev[2], 0.0f));
    atomicAdd(ap + 3, fmaxf(hv[3] + (float)ev[3], 0.0f));
}

// ---------------------------------------------------------------------------
// BN finalize: scale = g*rsqrt(var+eps), shift = beta - mean*scale
// ---------------------------------------------------------------------------
__global__ void k_bn_fin(const float* __restrict__ sum, const float* __restrict__ sq,
                         const float* __restrict__ g, const float* __restrict__ be,
                         float* __restrict__ sc, float* __restrict__ sh,
                         int C, float invN)
{
    int c = blockIdx.x * blockDim.x + threadIdx.x;
    if (c >= C) return;
    float mu  = sum[c] * invN;
    float var = sq[c] * invN - mu * mu;
    float rs  = rsqrtf(var + 1e-5f) * g[c];
    sc[c] = rs;
    sh[c] = be[c] - mu * rs;
}

// h = relu(v*scale + shift)
__global__ void k_hupd(const float* __restrict__ v, const float* __restrict__ sc,
                       const float* __restrict__ sh, float* __restrict__ h, long NH)
{
    long i = (long)blockIdx.x * 256 + threadIdx.x;
    if (i >= NH) return;
    int c = (int)(i & 127);
    h[i] = fmaxf(v[i] * sc[c] + sh[c], 0.0f);
}

// mean-pool part 1: atomic accumulate sums + counts, one wave per node
__global__ __launch_bounds__(256)
void k_pool(const float* __restrict__ h, const int* __restrict__ batch,
            float* __restrict__ psum, float* __restrict__ cnt, int N)
{
    int w = (blockIdx.x * 256 + threadIdx.x) >> 5;
    int lane = threadIdx.x & 31;
    if (w >= N) return;
    int b = batch[w];
    int c = lane * 4;
    v4f hv = *(const v4f*)(h + (long)w * 128 + c);
    float* pp = psum + (long)b * 128 + c;
    atomicAdd(pp + 0, hv[0]);
    atomicAdd(pp + 1, hv[1]);
    atomicAdd(pp + 2, hv[2]);
    atomicAdd(pp + 3, hv[3]);
    if (lane == 0) atomicAdd(&cnt[b], 1.0f);
}

__global__ void k_pooldiv(const float* __restrict__ psum, const float* __restrict__ cnt,
                          float* __restrict__ pooled, int GH)
{
    int i = blockIdx.x * 256 + threadIdx.x;
    if (i >= GH) return;
    int g = i >> 7;
    pooled[i] = psum[i] / fmaxf(cnt[g], 1.0f);
}

// ---------------------------------------------------------------------------
extern "C" void kernel_launch(void* const* d_in, const int* in_sizes, int n_in,
                              void* d_out, int out_size, void* d_ws, size_t ws_size,
                              hipStream_t stream)
{
    const float* x         = (const float*)d_in[0];
    const float* edge_attr = (const float*)d_in[1];
    const int*   edge_index= (const int*)  d_in[2];
    const int*   batch     = (const int*)  d_in[3];
    const float* node_W    = (const float*)d_in[4];
    const float* node_b    = (const float*)d_in[5];
    const float* edge_W    = (const float*)d_in[6];
    const float* edge_b    = (const float*)d_in[7];
    const float* W1s       = (const float*)d_in[8];
    const float* b1s       = (const float*)d_in[9];
    const float* g1s       = (const float*)d_in[10];
    const float* be1s      = (const float*)d_in[11];
    const float* W2s       = (const float*)d_in[12];
    const float* b2s       = (const float*)d_in[13];
    const float* epss      = (const float*)d_in[14];
    const float* bn_g      = (const float*)d_in[15];
    const float* bn_b      = (const float*)d_in[16];
    const float* ro_W1     = (const float*)d_in[17];
    const float* ro_b1     = (const float*)d_in[18];
    const float* ro_W2     = (const float*)d_in[19];
    const float* ro_b2     = (const float*)d_in[20];

    const int N = in_sizes[0] / 64;   // 100000
    const int E = in_sizes[1] / 16;   // 1600000
    const int G = 1024, H = 128;

    // ---- carve workspace ----
    char* wsp = (char*)d_ws;
    auto carve = [&](size_t bytes) -> void* {
        void* p = wsp;
        wsp += (bytes + 255) & ~(size_t)255;
        return p;
    };
    __bf16* e_bf   = (__bf16*)carve((size_t)E * 128 * 2);
    float*  hbuf   = (float*) carve((size_t)N * 128 * 4);
    float*  aggr   = (float*) carve((size_t)N * 128 * 4);
    __bf16* u_bf   = (__bf16*)carve((size_t)N * 256 * 2);
    float*  vbuf   = (float*) carve((size_t)N * 128 * 4);
    __bf16* wt_node= (__bf16*)carve(128 * 64 * 2);
    __bf16* wt_edge= (__bf16*)carve(128 * 32 * 2);
    __bf16* wt_w1  = (__bf16*)carve((size_t)3 * 256 * 128 * 2);
    __bf16* wt_w2  = (__bf16*)carve((size_t)3 * 128 * 256 * 2);
    __bf16* wt_ro1 = (__bf16*)carve(128 * 128 * 2);
    __bf16* wt_ro2 = (__bf16*)carve(128 * 128 * 2);
    float*  bn1s   = (float*) carve(2 * 256 * 4);     // sum | sumsq
    float*  bn1q   = bn1s + 256;
    float*  sc1    = (float*) carve(256 * 4);
    float*  sh1    = (float*) carve(256 * 4);
    float*  bn2s   = (float*) carve(2 * 128 * 4);
    float*  bn2q   = bn2s + 128;
    float*  sc2    = (float*) carve(128 * 4);
    float*  sh2    = (float*) carve(128 * 4);
    float*  psum   = (float*) carve((size_t)G * 128 * 4 + G * 4);  // sums | counts
    float*  cnt    = psum + (size_t)G * 128;
    float*  pooled = (float*) carve((size_t)G * 128 * 4);
    __bf16* t_bf   = (__bf16*)carve((size_t)G * 128 * 2);

    // ---- convert all weights to transposed bf16 ----
    k_cvt_wt<<<(128*64  + 255)/256, 256, 0, stream>>>(node_W, wt_node, 64, 64, 128);
    k_cvt_wt<<<(128*32  + 255)/256, 256, 0, stream>>>(edge_W, wt_edge, 16, 32, 128);
    for (int i = 0; i < 3; ++i) {
        k_cvt_wt<<<(256*128 + 255)/256, 256, 0, stream>>>(W1s + (size_t)i*128*256,
                                                          wt_w1 + (size_t)i*256*128, 128, 128, 256);
        k_cvt_wt<<<(128*256 + 255)/256, 256, 0, stream>>>(W2s + (size_t)i*256*128,
                                                          wt_w2 + (size_t)i*128*256, 256, 256, 128);
    }
    k_cvt_wt<<<(128*128 + 255)/256, 256, 0, stream>>>(ro_W1, wt_ro1, 128, 128, 128);
    k_cvt_wt<<<(128*128 + 255)/256, 256, 0, stream>>>(ro_W2, wt_ro2, 128, 128, 128);

    const unsigned gN = (unsigned)((N + 63) / 64);
    const unsigned gE = (unsigned)((E + 63) / 64);

    // ---- embeddings ----
    // h = x @ node_W + node_b  (fp32 out)
    k_gemm<64, 128, 0, 1><<<gN, 128, 0, stream>>>(x, nullptr, nullptr, nullptr, nullptr, 0,
                                                  wt_node, node_b, hbuf, nullptr,
                                                  nullptr, nullptr, N);
    // e = edge_attr @ edge_W + edge_b  (bf16 out, K padded 16->32)
    k_gemm<32, 128, 1, 2><<<gE, 128, 0, stream>>>(edge_attr, nullptr, nullptr, nullptr, nullptr, 0,
                                                  wt_edge, edge_b, nullptr, e_bf,
                                                  nullptr, nullptr, E);

    // ---- 3 GINE layers ----
    for (int i = 0; i < 3; ++i) {
        hipMemsetAsync(aggr, 0, (size_t)N * 128 * 4, stream);
        k_edge_msg<<<(unsigned)((E + 7) / 8), 256, 0, stream>>>(hbuf, e_bf, edge_index, aggr, E);

        hipMemsetAsync(bn1s, 0, 2 * 256 * 4, stream);
        // u = ((1+eps)h + aggr) @ W1 + b1  (bf16 out + BN stats)
        k_gemm<128, 256, 2, 2 | 8><<<gN, 128, 0, stream>>>(hbuf, aggr, nullptr, nullptr, epss, i,
                                                           wt_w1 + (size_t)i*256*128, b1s + i*256,
                                                           nullptr, u_bf, bn1s, bn1q, N);
        k_bn_fin<<<1, 256, 0, stream>>>(bn1s, bn1q, g1s + i*256, be1s + i*256,
                                        sc1, sh1, 256, 1.0f / (float)N);

        hipMemsetAsync(bn2s, 0, 2 * 128 * 4, stream);
        // v = relu(bn1(u)) @ W2 + b2  (fp32 out + BN stats)
        k_gemm<256, 128, 3, 1 | 8><<<gN, 128, 0, stream>>>(u_bf, nullptr, sc1, sh1, nullptr, 0,
                                                           wt_w2 + (size_t)i*128*256, b2s + i*128,
                                                           vbuf, nullptr, bn2s, bn2q, N);
        k_bn_fin<<<1, 128, 0, stream>>>(bn2s, bn2q, bn_g + i*128, bn_b + i*128,
                                        sc2, sh2, 128, 1.0f / (float)N);
        // h = relu(bn2(v))
        long NH = (long)N * 128;
        k_hupd<<<(unsigned)((NH + 255) / 256), 256, 0, stream>>>(vbuf, sc2, sh2, hbuf, NH);
    }

    // ---- global mean pool ----
    hipMemsetAsync(psum, 0, (size_t)G * 128 * 4 + G * 4, stream);
    k_pool<<<(unsigned)((N + 7) / 8), 256, 0, stream>>>(hbuf, batch, psum, cnt, N);
    k_pooldiv<<<(G * H + 255) / 256, 256, 0, stream>>>(psum, cnt, pooled, G * H);

    // ---- readout MLP ----
    const unsigned gG = (unsigned)(G / 64);
    k_gemm<128, 128, 0, 2 | 4><<<gG, 128, 0, stream>>>(pooled, nullptr, nullptr, nullptr, nullptr, 0,
                                                       wt_ro1, ro_b1, nullptr, t_bf,
                                                       nullptr, nullptr, G);
    k_gemm<128, 128, 4, 1><<<gG, 128, 0, stream>>>(t_bf, nullptr, nullptr, nullptr, nullptr, 0,
                                                   wt_ro2, ro_b2, (float*)d_out, nullptr,
                                                   nullptr, nullptr, G);
}
